// TriangleAttentionEndingNode_7215545057298
// MI455X (gfx1250) — matrix operand order repaired and checked by reading, not compiled, not run on hardware
//
#include <hip/hip_runtime.h>
#include <hip/hip_bf16.h>
#include <math.h>

typedef _Float16 v16h __attribute__((ext_vector_type(16)));
typedef _Float16 v8h  __attribute__((ext_vector_type(8)));
typedef float    v8f  __attribute__((ext_vector_type(8)));

#define NRES 256
#define CCH  128
#define NH   4
#define CHD  32

// LDS row strides (in halfs) — multiples of 8 (16B alignment), bank-conflict friendly
#define RS_X  136
#define RS_Q  40
#define RS_VT 264
#define RS_P  264
#define RS_O  136
#define RS_W  136

// LDS byte offsets (all 16B aligned)
#define OFF_X    0          // 256*136*2 = 69632
#define OFF_Q    69632      // 256*40*2  = 20480
#define OFF_K    90112      // 20480
#define OFF_G    110592     // 20480
#define OFF_VT   131072     // 32*264*2  = 16896
#define OFF_OALL 147968     // 69632
#define OFF_WST  217600     // 32*136*2  = 8704
#define OFF_P    226304     // 8*16*264*2 = 67584
#define OFF_MB   293888     // 1024
#define OFF_LNG  294912     // 512
#define OFF_LNB  295424     // 512
#define SMEM_BYTES 295936

__device__ __forceinline__ v8f vzero() {
  v8f z;
#pragma unroll
  for (int t = 0; t < 8; ++t) z[t] = 0.0f;
  return z;
}

__device__ __forceinline__ v8f wmma16(v16h a, v16h b, v8f c) {
  return __builtin_amdgcn_wmma_f32_16x16x32_f16(false, a, false, b, (short)0, c,
                                                false, false);
}

// A matrix 16x32 f16: lane (i0,hi) holds row i0, k = hi*8+[0..7] and 16+hi*8+[0..7]
__device__ __forceinline__ v16h load_a_frag(const _Float16* base, int rs, int row0, int k0) {
  const int lane = threadIdx.x & 31;
  const int i0 = lane & 15, hi = lane >> 4;
  const _Float16* p = base + (size_t)(row0 + i0) * rs + k0 + hi * 8;
  v8h lo = *(const v8h*)p;
  v8h hc = *(const v8h*)(p + 16);
  v16h r;
#pragma unroll
  for (int t = 0; t < 8; ++t) { r[t] = lo[t]; r[t + 8] = hc[t]; }
  return r;
}

// B matrix 32x16 f16: lane (i0,hi) holds column i0, k = hi*8+[0..7] and 16+hi*8+[0..7]
__device__ __forceinline__ v16h load_b_frag(const _Float16* base, int rs, int n0, int k0) {
  const int lane = threadIdx.x & 31;
  const int i0 = lane & 15, hi = lane >> 4;
  const _Float16* p = base + (size_t)(n0 + i0) * rs + k0 + hi * 8;
  v8h lo = *(const v8h*)p;
  v8h hc = *(const v8h*)(p + 16);
  v16h r;
#pragma unroll
  for (int t = 0; t < 8; ++t) { r[t] = lo[t]; r[t + 8] = hc[t]; }
  return r;
}

// ---------------------------------------------------------------------------
// Kernel 1: pair bias pb[h][i][j] = dot(LN(act[0,j,i,:]), w2d[:,h])
// block r1 = j, waves iterate r2 = i
// ---------------------------------------------------------------------------
__global__ __launch_bounds__(256) void pair_bias_kernel(
    const float* __restrict__ act, const float* __restrict__ ln_g,
    const float* __restrict__ ln_b, const float* __restrict__ w2d,
    float* __restrict__ pb) {
  __shared__ float w2ds[512];
  __shared__ float lng[128], lnb[128];
  const int tid = threadIdx.x, lane = tid & 31, w = tid >> 5;
  const int r1 = blockIdx.x;
  if (tid < 128) { lng[tid] = ln_g[tid]; lnb[tid] = ln_b[tid]; }
  for (int e = tid; e < 512; e += 256) w2ds[e] = w2d[e];
  __syncthreads();
  for (int itr = 0; itr < 32; ++itr) {
    const int r2 = itr * 8 + w;
    const float4 v = ((const float4*)(act + ((size_t)r1 * NRES + r2) * CCH))[lane];
    float s = v.x + v.y + v.z + v.w;
    float ss = v.x * v.x + v.y * v.y + v.z * v.z + v.w * v.w;
#pragma unroll
    for (int d = 1; d < 32; d <<= 1) { s += __shfl_xor(s, d, 32); ss += __shfl_xor(ss, d, 32); }
    const float mean = s * (1.0f / 128.0f);
    const float rstd = rsqrtf(ss * (1.0f / 128.0f) - mean * mean + 1e-5f);
    const int c = lane * 4;
    float y[4];
    y[0] = (v.x - mean) * rstd * lng[c + 0] + lnb[c + 0];
    y[1] = (v.y - mean) * rstd * lng[c + 1] + lnb[c + 1];
    y[2] = (v.z - mean) * rstd * lng[c + 2] + lnb[c + 2];
    y[3] = (v.w - mean) * rstd * lng[c + 3] + lnb[c + 3];
    float p0 = 0.f, p1 = 0.f, p2 = 0.f, p3 = 0.f;
#pragma unroll
    for (int t = 0; t < 4; ++t) {
      p0 += y[t] * w2ds[(c + t) * 4 + 0];
      p1 += y[t] * w2ds[(c + t) * 4 + 1];
      p2 += y[t] * w2ds[(c + t) * 4 + 2];
      p3 += y[t] * w2ds[(c + t) * 4 + 3];
    }
#pragma unroll
    for (int d = 1; d < 32; d <<= 1) {
      p0 += __shfl_xor(p0, d, 32); p1 += __shfl_xor(p1, d, 32);
      p2 += __shfl_xor(p2, d, 32); p3 += __shfl_xor(p3, d, 32);
    }
    if (lane == 0) {
      pb[((size_t)0 * NRES + r2) * NRES + r1] = p0;
      pb[((size_t)1 * NRES + r2) * NRES + r1] = p1;
      pb[((size_t)2 * NRES + r2) * NRES + r1] = p2;
      pb[((size_t)3 * NRES + r2) * NRES + r1] = p3;
    }
  }
}

// ---------------------------------------------------------------------------
// Kernel 2: one block per m. 256 threads = 8 wave32.
// ---------------------------------------------------------------------------
__global__ __launch_bounds__(256) void tri_attn_kernel(
    const float* __restrict__ act, const float* __restrict__ mask,
    const float* __restrict__ ln_g, const float* __restrict__ ln_b,
    const float* __restrict__ wq, const float* __restrict__ wk,
    const float* __restrict__ wv, const float* __restrict__ wg,
    const float* __restrict__ bg, const float* __restrict__ wo,
    const float* __restrict__ bo, const float* __restrict__ pb,
    float* __restrict__ out) {
  extern __shared__ char smem[];
  _Float16* X   = (_Float16*)(smem + OFF_X);
  _Float16* Q   = (_Float16*)(smem + OFF_Q);
  _Float16* K   = (_Float16*)(smem + OFF_K);
  _Float16* G   = (_Float16*)(smem + OFF_G);
  _Float16* Vt  = (_Float16*)(smem + OFF_VT);
  _Float16* Oa  = (_Float16*)(smem + OFF_OALL);
  _Float16* Wst = (_Float16*)(smem + OFF_WST);
  _Float16* Pf  = (_Float16*)(smem + OFF_P);
  float* mb  = (float*)(smem + OFF_MB);
  float* lng = (float*)(smem + OFF_LNG);
  float* lnb = (float*)(smem + OFF_LNB);

  const int tid = threadIdx.x;
  const int lane = tid & 31;
  const int w = tid >> 5;
  const int i0 = lane & 15;
  const int hi = lane >> 4;
  const int m = blockIdx.x;

  if (tid < 128) { lng[tid] = ln_g[tid]; lnb[tid] = ln_b[tid]; }
  // additive mask bias over keys j: 1e9*(mask[0,j,m]-1)
  mb[tid] = 1e9f * (mask[(size_t)tid * NRES + m] - 1.0f);

  // LayerNorm: X[i][:] = LN(act[0, i, m, :])  (one wave per row, float4/lane)
  for (int it = 0; it < 32; ++it) {
    const int i = it * 8 + w;
    const float4 v = ((const float4*)(act + ((size_t)i * NRES + m) * CCH))[lane];
    float s = v.x + v.y + v.z + v.w;
    float ss = v.x * v.x + v.y * v.y + v.z * v.z + v.w * v.w;
#pragma unroll
    for (int d = 1; d < 32; d <<= 1) { s += __shfl_xor(s, d, 32); ss += __shfl_xor(ss, d, 32); }
    const float mean = s * (1.0f / 128.0f);
    const float rstd = rsqrtf(ss * (1.0f / 128.0f) - mean * mean + 1e-5f);
    const int c = lane * 4;
    _Float16* xr = X + (size_t)i * RS_X + c;
    xr[0] = (_Float16)((v.x - mean) * rstd * lng[c + 0] + lnb[c + 0]);
    xr[1] = (_Float16)((v.y - mean) * rstd * lng[c + 1] + lnb[c + 1]);
    xr[2] = (_Float16)((v.z - mean) * rstd * lng[c + 2] + lnb[c + 2]);
    xr[3] = (_Float16)((v.w - mean) * rstd * lng[c + 3] + lnb[c + 3]);
  }
  __syncthreads();

  for (int h = 0; h < NH; ++h) {
    // ------------------ Q projection (fold 1/sqrt(CH) into Q) ------------------
    for (int e = tid; e < 4096; e += 256) {
      const int o = e & 31, c = e >> 5;
      Wst[o * RS_W + c] = (_Float16)wq[(size_t)c * CCH + h * CHD + o];
    }
    __syncthreads();
    for (int job = w; job < 32; job += 8) {
      const int mt = job >> 1, nt = job & 1;
      v8f acc = vzero();
#pragma unroll
      for (int ks = 0; ks < 4; ++ks)
        acc = wmma16(load_a_frag(X, RS_X, mt * 16, ks * 32),
                     load_b_frag(Wst, RS_W, nt * 16, ks * 32), acc);
#pragma unroll
      for (int r = 0; r < 8; ++r)
        Q[(size_t)(mt * 16 + hi * 8 + r) * RS_Q + nt * 16 + i0] =
            (_Float16)(acc[r] * 0.17677669529663687f);
    }
    __syncthreads();
    // ------------------ K projection ------------------
    for (int e = tid; e < 4096; e += 256) {
      const int o = e & 31, c = e >> 5;
      Wst[o * RS_W + c] = (_Float16)wk[(size_t)c * CCH + h * CHD + o];
    }
    __syncthreads();
    for (int job = w; job < 32; job += 8) {
      const int mt = job >> 1, nt = job & 1;
      v8f acc = vzero();
#pragma unroll
      for (int ks = 0; ks < 4; ++ks)
        acc = wmma16(load_a_frag(X, RS_X, mt * 16, ks * 32),
                     load_b_frag(Wst, RS_W, nt * 16, ks * 32), acc);
#pragma unroll
      for (int r = 0; r < 8; ++r)
        K[(size_t)(mt * 16 + hi * 8 + r) * RS_Q + nt * 16 + i0] = (_Float16)acc[r];
    }
    __syncthreads();
    // ------------------ V projection (store transposed Vt[c][j]) ------------------
    for (int e = tid; e < 4096; e += 256) {
      const int o = e & 31, c = e >> 5;
      Wst[o * RS_W + c] = (_Float16)wv[(size_t)c * CCH + h * CHD + o];
    }
    __syncthreads();
    for (int job = w; job < 32; job += 8) {
      const int mt = job >> 1, nt = job & 1;
      v8f acc = vzero();
#pragma unroll
      for (int ks = 0; ks < 4; ++ks)
        acc = wmma16(load_a_frag(X, RS_X, mt * 16, ks * 32),
                     load_b_frag(Wst, RS_W, nt * 16, ks * 32), acc);
      v8h hv;
#pragma unroll
      for (int r = 0; r < 8; ++r) hv[r] = (_Float16)acc[r];
      *(v8h*)(Vt + (size_t)(nt * 16 + i0) * RS_VT + mt * 16 + hi * 8) = hv;
    }
    __syncthreads();
    // ------------------ Gate: sigmoid(X@wg + bg) ------------------
    for (int e = tid; e < 4096; e += 256) {
      const int o = e & 31, c = e >> 5;
      Wst[o * RS_W + c] = (_Float16)wg[(size_t)c * CCH + h * CHD + o];
    }
    __syncthreads();
    for (int job = w; job < 32; job += 8) {
      const int mt = job >> 1, nt = job & 1;
      v8f acc = vzero();
#pragma unroll
      for (int ks = 0; ks < 4; ++ks)
        acc = wmma16(load_a_frag(X, RS_X, mt * 16, ks * 32),
                     load_b_frag(Wst, RS_W, nt * 16, ks * 32), acc);
      const float bias = bg[h * CHD + nt * 16 + i0];
#pragma unroll
      for (int r = 0; r < 8; ++r) {
        const float gv = 1.0f / (1.0f + __expf(-(acc[r] + bias)));
        G[(size_t)(mt * 16 + hi * 8 + r) * RS_Q + nt * 16 + i0] = (_Float16)gv;
      }
    }
    __syncthreads();

    // ------------------ Attention: S^T = K * Q^T, softmax over j, O^T = Vt * P^T -----
    _Float16* pw = Pf + (size_t)w * 16 * RS_P;  // per-wave P tile [16 i][256 j]
#pragma unroll 1
    for (int pass = 0; pass < 2; ++pass) {
      const int it = w + pass * 8;
      const int ibase = it * 16;
      const v16h bq = load_b_frag(Q, RS_Q, ibase, 0);  // B: n=i, k=c (CH=32 = one WMMA K)
      v8f S[16];                                        // S[jt] holds (M=j, N=i)
#pragma unroll
      for (int jt = 0; jt < 16; ++jt)
        S[jt] = wmma16(load_a_frag(K, RS_Q, jt * 16, 0), bq, vzero());
      // add pair bias + mask bias; lane holds column i = ibase+i0, rows j = jt*16+hi*8+r
      const int icol = ibase + i0;
      const float* pbrow = pb + ((size_t)h * NRES + icol) * NRES;
#pragma unroll
      for (int jt = 0; jt < 16; ++jt)
#pragma unroll
        for (int r = 0; r < 8; ++r) {
          const int j = jt * 16 + hi * 8 + r;
          S[jt][r] += pbrow[j] + mb[j];
        }
      // softmax over j: local reduce over 128 regs, then combine with lane^16
      float mx = -3.0e38f;
#pragma unroll
      for (int jt = 0; jt < 16; ++jt)
#pragma unroll
        for (int r = 0; r < 8; ++r) mx = fmaxf(mx, S[jt][r]);
      mx = fmaxf(mx, __shfl_xor(mx, 16, 32));
      float sum = 0.0f;
#pragma unroll
      for (int jt = 0; jt < 16; ++jt)
#pragma unroll
        for (int r = 0; r < 8; ++r) {
          const float e = __expf(S[jt][r] - mx);
          S[jt][r] = e;
          sum += e;
        }
      sum += __shfl_xor(sum, 16, 32);
      const float inv = 1.0f / sum;
      // store P row-major: lane's 8 values are j-consecutive -> one b128 per jt
#pragma unroll
      for (int jt = 0; jt < 16; ++jt) {
        v8h hp;
#pragma unroll
        for (int r = 0; r < 8; ++r) hp[r] = (_Float16)(S[jt][r] * inv);
        *(v8h*)(pw + (size_t)i0 * RS_P + jt * 16 + hi * 8) = hp;
      }
      // O^T[c][i] = sum_j Vt[c][j] * P[i][j]; gate and write into Oa[i][h*32+c]
      const int ig = ibase + i0;
#pragma unroll
      for (int ct = 0; ct < 2; ++ct) {
        v8f oa = vzero();
#pragma unroll
        for (int kt = 0; kt < 8; ++kt)
          oa = wmma16(load_a_frag(Vt, RS_VT, ct * 16, kt * 32),
                      load_b_frag(pw, RS_P, 0, kt * 32), oa);
        v8h ho;
#pragma unroll
        for (int r = 0; r < 8; ++r) {
          const int c = ct * 16 + hi * 8 + r;
          const float gv = (float)G[(size_t)ig * RS_Q + c];
          ho[r] = (_Float16)(oa[r] * gv);
        }
        *(v8h*)(Oa + (size_t)ig * RS_O + h * CHD + ct * 16 + hi * 8) = ho;
      }
    }
    __syncthreads();  // before next head restages weights / Q/K/Vt/G
  }

  // ------------------ Output projection: out = Oa @ wo + bo, stored transposed -----
  _Float16* WOT = (_Float16*)(smem + OFF_Q);  // reuse Q+K region: 128*136 halfs
  for (int e = tid; e < 16384; e += 256) {
    const int o = e & 127, c = e >> 7;
    WOT[(size_t)o * RS_W + c] = (_Float16)wo[(size_t)c * CCH + o];
  }
  __syncthreads();
  for (int job = w; job < 128; job += 8) {
    const int mt = job >> 3, nt = job & 7;
    v8f acc = vzero();
#pragma unroll
    for (int ks = 0; ks < 4; ++ks)
      acc = wmma16(load_a_frag(Oa, RS_O, mt * 16, ks * 32),
                   load_b_frag(WOT, RS_W, nt * 16, ks * 32), acc);
    const int o = nt * 16 + i0;
    const float bias = bo[o];
#pragma unroll
    for (int r = 0; r < 8; ++r) {
      const int i = mt * 16 + hi * 8 + r;
      out[((size_t)i * NRES + m) * CCH + o] = acc[r] + bias;  // final[b, i, m, :]
    }
  }
}

extern "C" void kernel_launch(void* const* d_in, const int* in_sizes, int n_in,
                              void* d_out, int out_size, void* d_ws, size_t ws_size,
                              hipStream_t stream) {
  (void)in_sizes; (void)n_in; (void)out_size; (void)ws_size;
  const float* act  = (const float*)d_in[0];
  const float* mask = (const float*)d_in[1];
  const float* ln_g = (const float*)d_in[2];
  const float* ln_b = (const float*)d_in[3];
  const float* wq   = (const float*)d_in[4];
  const float* wk   = (const float*)d_in[5];
  const float* wv   = (const float*)d_in[6];
  const float* w2d  = (const float*)d_in[7];
  const float* wg   = (const float*)d_in[8];
  const float* bg   = (const float*)d_in[9];
  const float* wo   = (const float*)d_in[10];
  const float* bo   = (const float*)d_in[11];
  float* out = (float*)d_out;
  float* pb  = (float*)d_ws;  // 4*256*256 f32 = 1 MB scratch (L2 resident)

  hipFuncSetAttribute(reinterpret_cast<const void*>(tri_attn_kernel),
                      hipFuncAttributeMaxDynamicSharedMemorySize, SMEM_BYTES);

  pair_bias_kernel<<<NRES, 256, 0, stream>>>(act, ln_g, ln_b, w2d, pb);
  tri_attn_kernel<<<NRES, 256, SMEM_BYTES, stream>>>(
      act, mask, ln_g, ln_b, wq, wk, wv, wg, bg, wo, bo, pb, out);
}
